// TangentSpaceLinearAttention_16870631539364
// MI455X (gfx1250) — matrix operand order repaired and checked by reading, not compiled, not run on hardware
//
#include <hip/hip_runtime.h>
#include <hip/hip_bf16.h>
#include <math.h>

// ---------------------------------------------------------------- types
typedef __attribute__((ext_vector_type(16))) __bf16 v16bf;
typedef __attribute__((ext_vector_type(8)))  float  v8f;
typedef __attribute__((ext_vector_type(4)))  unsigned int u32x4;

union FragAB { v16bf v; u32x4 q[2]; };

#define Bb 4
#define Nn 4096
#define Dd 1024
#define Hh 16
#define DHh 64
#define NCHUNK 8
#define CHUNKN (Nn / NCHUNK)     // 512

__device__ __forceinline__ unsigned short f2bf(float f) {
    unsigned int u = __float_as_uint(f);
    unsigned int r = u + 0x7FFFu + ((u >> 16) & 1u);   // RNE
    return (unsigned short)(r >> 16);
}
__device__ __forceinline__ float bf2f(unsigned short u) {
    return __uint_as_float(((unsigned int)u) << 16);
}

// 16B global -> LDS via the CDNA5 async-LDS path (ASYNCcnt tracked, no data VGPRs).
// LDS byte offset = low 32 bits of the generic pointer.
__device__ __forceinline__ void async_copy16(const unsigned short* g, unsigned short* l) {
    unsigned int loff = (unsigned int)(size_t)l;
    asm volatile("global_load_async_to_lds_b128 %0, %1, off"
                 :: "v"(loff), "v"(g) : "memory");
}
__device__ __forceinline__ void wait_async0() {
    asm volatile("s_wait_asynccnt 0" ::: "memory");
}

// ---------------------------------------------------------------- stage 0: fp32 -> bf16 weights
__global__ __launch_bounds__(256) void cvt_bf16_kernel(const float* __restrict__ src,
                                                       unsigned short* __restrict__ dst, int n) {
    int i = blockIdx.x * 256 + threadIdx.x;
    if (i < n) dst[i] = f2bf(src[i]);
}

// ---------------------------------------------------------------- stage 1: log map at origin
__global__ __launch_bounds__(256) void logmap_kernel(const float* __restrict__ x,
                                                     const float* __restrict__ curv,
                                                     unsigned short* __restrict__ xt) {
    const int t = blockIdx.x;                       // token in [0, B*N)
    const float* row = x + (size_t)t * Dd;
    __shared__ float red[256];
    float ss = 0.f;
    for (int i = threadIdx.x; i < Dd; i += 256) { float v = row[i]; ss += v * v; }
    red[threadIdx.x] = ss; __syncthreads();
    for (int s = 128; s > 0; s >>= 1) {
        if (threadIdx.x < s) red[threadIdx.x] += red[threadIdx.x + s];
        __syncthreads();
    }
    float c  = fmaxf(fabsf(curv[0]), 1e-6f);
    float sc = sqrtf(fmaxf(c, 1e-6f));
    float nrm = fminf(fmaxf(sqrtf(red[0]), 1e-6f), 0.999f);
    float a = sc * nrm;
    float scale = 0.5f * __logf((1.f + a) / (1.f - a)) / (a + 1e-8f);   // arctanh(a)/a'
    unsigned short* orow = xt + (size_t)t * Dd;
    for (int i = threadIdx.x; i < Dd; i += 256) orow[i] = f2bf(row[i] * scale);
}

// ---------------------------------------------------------------- stage 2/5: bf16 WMMA GEMM
// C[m][n] = sum_k A[m][k] * Bw[n][k] + bias[n]   (nn.Linear: X @ W^T + b)
// MODE 0: store bf16 with phi = elu+1 ; MODE 1: store bf16 ; MODE 2: store f32
// 128x128x32 tile, 8 waves (each 64x32), double-buffered LDS filled by async-LDS loads.
template <int MODE>
__global__ __launch_bounds__(256) void gemm_bf16_kernel(const unsigned short* __restrict__ A,
                                                        const unsigned short* __restrict__ Bw,
                                                        const float* __restrict__ bias,
                                                        unsigned short* __restrict__ Cb,
                                                        float* __restrict__ Cf,
                                                        int M, int Nc, int K) {
    __shared__ __attribute__((aligned(16))) unsigned short As[2][128 * 40]; // row stride 80B
    __shared__ __attribute__((aligned(16))) unsigned short Bs[2][128 * 40];

    const int tid   = threadIdx.x;
    const int lane  = tid & 31;
    const int wid   = tid >> 5;      // 0..7
    const int waveM = wid & 1;       // 2 x 64 rows
    const int waveN = wid >> 1;      // 4 x 32 cols
    const int half  = lane >> 4;
    const int l15   = lane & 15;
    const int m0 = blockIdx.y * 128;
    const int n0 = blockIdx.x * 128;

    // this thread's two staging slots (row, seg of 8 bf16)
    const int idx0 = tid * 2, idx1 = tid * 2 + 1;
    const int rowS0 = idx0 >> 2, segS0 = idx0 & 3;
    const int rowS1 = idx1 >> 2, segS1 = idx1 & 3;

    v8f acc[4][2];
#pragma unroll
    for (int i = 0; i < 4; i++)
#pragma unroll
        for (int j = 0; j < 2; j++)
#pragma unroll
            for (int r = 0; r < 8; r++) acc[i][j][r] = 0.f;

    const int nsteps = K / 32;
    int cur = 0;

    // prologue: stage k-step 0 into buffer 0
    async_copy16(A  + (size_t)(m0 + rowS0) * K + segS0 * 8, &As[0][rowS0 * 40 + segS0 * 8]);
    async_copy16(Bw + (size_t)(n0 + rowS0) * K + segS0 * 8, &Bs[0][rowS0 * 40 + segS0 * 8]);
    async_copy16(A  + (size_t)(m0 + rowS1) * K + segS1 * 8, &As[0][rowS1 * 40 + segS1 * 8]);
    async_copy16(Bw + (size_t)(n0 + rowS1) * K + segS1 * 8, &Bs[0][rowS1 * 40 + segS1 * 8]);
    wait_async0();
    __syncthreads();

    for (int s = 0; s < nsteps; ++s) {
        // prefetch next k-step into the other buffer (async engine runs under WMMAs)
        if (s + 1 < nsteps) {
            const int k1 = (s + 1) * 32;
            const int nxt = cur ^ 1;
            async_copy16(A  + (size_t)(m0 + rowS0) * K + k1 + segS0 * 8,
                         &As[nxt][rowS0 * 40 + segS0 * 8]);
            async_copy16(Bw + (size_t)(n0 + rowS0) * K + k1 + segS0 * 8,
                         &Bs[nxt][rowS0 * 40 + segS0 * 8]);
            async_copy16(A  + (size_t)(m0 + rowS1) * K + k1 + segS1 * 8,
                         &As[nxt][rowS1 * 40 + segS1 * 8]);
            async_copy16(Bw + (size_t)(n0 + rowS1) * K + k1 + segS1 * 8,
                         &Bs[nxt][rowS1 * 40 + segS1 * 8]);
        }

        // compute from current buffer
        FragAB bfr[2];
#pragma unroll
        for (int j = 0; j < 2; j++) {               // B frag: col=l15, K = half*16 + e
            const unsigned short* p = &Bs[cur][(waveN * 32 + j * 16 + l15) * 40 + half * 16];
            bfr[j].q[0] = *(const u32x4*)(p);
            bfr[j].q[1] = *(const u32x4*)(p + 8);
        }
#pragma unroll
        for (int i = 0; i < 4; i++) {               // A frag: row=l15, K = half*16 + e
            FragAB a;
            const unsigned short* p = &As[cur][(waveM * 64 + i * 16 + l15) * 40];
            a.q[0] = *(const u32x4*)(p + half * 16);
            a.q[1] = *(const u32x4*)(p + half * 16 + 8);
#pragma unroll
            for (int j = 0; j < 2; j++)
                acc[i][j] = __builtin_amdgcn_wmma_f32_16x16x32_bf16(
                    false, a.v, false, bfr[j].v, (short)0, acc[i][j], false, false);
        }

        if (s + 1 < nsteps) {
            wait_async0();        // own async writes to next buffer landed
            __syncthreads();      // everyone done reading cur + done filling next
            cur ^= 1;
        }
    }

#pragma unroll
    for (int i = 0; i < 4; i++)
#pragma unroll
        for (int j = 0; j < 2; j++) {
            int gn = n0 + waveN * 32 + j * 16 + l15;
            float bi = bias[gn];
#pragma unroll
            for (int r = 0; r < 8; r++) {
                int gm = m0 + waveM * 64 + i * 16 + r + half * 8;   // C: M = r + 8*half
                float v = acc[i][j][r] + bi;
                if (MODE == 0) v = (v > 0.f) ? (v + 1.f) : __expf(v); // elu(v)+1
                if (MODE == 2) Cf[(size_t)gm * Nc + gn] = v;
                else           Cb[(size_t)gm * Nc + gn] = f2bf(v);
            }
        }
}

// ---------------------------------------------------------------- k_sum partials (split-K, deterministic)
__global__ __launch_bounds__(64) void ksum_split_kernel(const unsigned short* __restrict__ Kp,
                                                        float* __restrict__ kspp) {
    int bh = blockIdx.x, chunk = blockIdx.y, f = threadIdx.x;
    int b = bh >> 4, h = bh & 15;
    const unsigned short* base = Kp + (size_t)b * Nn * Dd + h * DHh + f;
    float s = 0.f;
    for (int n = chunk * CHUNKN; n < (chunk + 1) * CHUNKN; n++) s += bf2f(base[(size_t)n * Dd]);
    kspp[(size_t)chunk * (Bb * Hh * DHh) + bh * DHh + f] = s;
}

// ---------------------------------------------------------------- stage 3: kv partials
// kv[f][d] = sum_n phi_k[n][f] v[n][d], split over NCHUNK n-ranges (deterministic partials)
__global__ __launch_bounds__(128) void kv_split_kernel(const unsigned short* __restrict__ Kp,
                                                       const unsigned short* __restrict__ Vp,
                                                       float* __restrict__ kvp) {
    __shared__ __attribute__((aligned(16))) unsigned short Kt[64 * 40];  // A^T: Kt[f][n_local]
    __shared__ __attribute__((aligned(16))) unsigned short Vt[64 * 40];  // B^T: Vt[d][n_local]
    const int bh = blockIdx.x, chunk = blockIdx.y;
    const int b = bh >> 4, h = bh & 15;
    const int tid = threadIdx.x, lane = tid & 31, w = tid >> 5;
    const int half = lane >> 4, l15 = lane & 15;
    const size_t rowBase = (size_t)b * Nn * Dd + h * DHh;

    v8f acc[4];
#pragma unroll
    for (int j = 0; j < 4; j++)
#pragma unroll
        for (int r = 0; r < 8; r++) acc[j][r] = 0.f;

    for (int n0 = chunk * CHUNKN; n0 < (chunk + 1) * CHUNKN; n0 += 32) {
        for (int it = 0; it < 2; ++it) {          // 32 rows x 64 cols, transpose into LDS
            int idx = tid * 2 + it;               // 0..255
            int row = idx >> 3, seg = idx & 7;
            union { u32x4 q; unsigned short u[8]; } uk, uv;
            uk.q = *(const u32x4*)(Kp + rowBase + (size_t)(n0 + row) * Dd + seg * 8);
            uv.q = *(const u32x4*)(Vp + rowBase + (size_t)(n0 + row) * Dd + seg * 8);
#pragma unroll
            for (int e = 0; e < 8; e++) {
                Kt[(seg * 8 + e) * 40 + row] = uk.u[e];
                Vt[(seg * 8 + e) * 40 + row] = uv.u[e];
            }
        }
        __syncthreads();

        FragAB a;
        const unsigned short* pa = Kt + (w * 16 + l15) * 40;
        a.q[0] = *(const u32x4*)(pa + half * 16);
        a.q[1] = *(const u32x4*)(pa + half * 16 + 8);
#pragma unroll
        for (int j = 0; j < 4; j++) {
            FragAB bfr;
            const unsigned short* pb = Vt + (j * 16 + l15) * 40 + half * 16;
            bfr.q[0] = *(const u32x4*)(pb);
            bfr.q[1] = *(const u32x4*)(pb + 8);
            acc[j] = __builtin_amdgcn_wmma_f32_16x16x32_bf16(
                false, a.v, false, bfr.v, (short)0, acc[j], false, false);
        }
        __syncthreads();
    }
    float* outp = kvp + (size_t)chunk * (Bb * Hh * DHh * DHh) + (size_t)bh * DHh * DHh;
#pragma unroll
    for (int j = 0; j < 4; j++) {
        int d = j * 16 + l15;
#pragma unroll
        for (int r = 0; r < 8; r++) {
            int f = w * 16 + r + half * 8;
            outp[f * DHh + d] = acc[j][r];
        }
    }
}

// ---------------------------------------------------------------- reduce partials -> kv(bf16), ksum(f32)
__global__ __launch_bounds__(256) void kv_reduce_kernel(const float* __restrict__ kvp,
                                                        const float* __restrict__ kspp,
                                                        unsigned short* __restrict__ KV,
                                                        float* __restrict__ ksum) {
    const int nkv = Bb * Hh * DHh * DHh;     // 262144
    const int nks = Bb * Hh * DHh;           // 4096
    int i = blockIdx.x * 256 + threadIdx.x;
    if (i < nkv) {
        float s = 0.f;
#pragma unroll
        for (int c = 0; c < NCHUNK; c++) s += kvp[(size_t)c * nkv + i];
        KV[i] = f2bf(s);
    } else if (i < nkv + nks) {
        int j = i - nkv;
        float s = 0.f;
#pragma unroll
        for (int c = 0; c < NCHUNK; c++) s += kspp[(size_t)c * nks + j];
        ksum[j] = s;
    }
}

// ---------------------------------------------------------------- stage 4: qkv + normalizer
__global__ __launch_bounds__(128) void qkv_kernel(const unsigned short* __restrict__ Qp,
                                                  const unsigned short* __restrict__ KV,
                                                  const float* __restrict__ Ksum,
                                                  unsigned short* __restrict__ Out) {
    __shared__ __attribute__((aligned(16))) unsigned short Qs[64 * 80];   // [m][f], stride 160B
    __shared__ __attribute__((aligned(16))) unsigned short KVt[64 * 80];  // [d][f]
    __shared__ float ks[64];
    __shared__ float nrm[64];
    const int bh = blockIdx.x, mt = blockIdx.y;
    const int b = bh >> 4, h = bh & 15;
    const int tid = threadIdx.x, lane = tid & 31, w = tid >> 5;
    const int half = lane >> 4, l15 = lane & 15;
    const size_t qbase = ((size_t)b * Nn + mt * 64) * Dd + h * DHh;

    for (int it = 0; it < 4; ++it) {               // stage phi_q tile (64x64)
        int idx = tid * 4 + it;                    // 0..511
        int row = idx >> 3, seg = idx & 7;
        *(u32x4*)(Qs + row * 80 + seg * 8) =
            *(const u32x4*)(Qp + qbase + (size_t)row * Dd + seg * 8);
    }
    for (int it = 0; it < 4; ++it) {               // stage kv transposed (B[f][d] -> KVt[d][f])
        int idx = tid * 4 + it;
        int f = idx >> 3, seg = idx & 7;
        union { u32x4 q; unsigned short u[8]; } u1;
        u1.q = *(const u32x4*)(KV + ((size_t)bh * DHh + f) * DHh + seg * 8);
#pragma unroll
        for (int e = 0; e < 8; e++) KVt[(seg * 8 + e) * 80 + f] = u1.u[e];
    }
    if (tid < 64) ks[tid] = Ksum[bh * DHh + tid];
    __syncthreads();

    if (tid < 64) {                                // normalizer = phi_q . k_sum (clamped)
        float s = 0.f;
        for (int f = 0; f < 64; f++) s += bf2f(Qs[tid * 80 + f]) * ks[f];
        nrm[tid] = fmaxf(s, 1e-6f);
    }
    __syncthreads();

    v8f acc[4];
#pragma unroll
    for (int j = 0; j < 4; j++)
#pragma unroll
        for (int r = 0; r < 8; r++) acc[j][r] = 0.f;
#pragma unroll
    for (int kk = 0; kk < 2; ++kk) {               // K = 64 -> two 32-steps
        FragAB a;
        const unsigned short* pa = Qs + (w * 16 + l15) * 80 + kk * 32;
        a.q[0] = *(const u32x4*)(pa + half * 16);
        a.q[1] = *(const u32x4*)(pa + half * 16 + 8);
#pragma unroll
        for (int j = 0; j < 4; j++) {
            FragAB bfr;
            const unsigned short* pb = KVt + (j * 16 + l15) * 80 + kk * 32 + half * 16;
            bfr.q[0] = *(const u32x4*)(pb);
            bfr.q[1] = *(const u32x4*)(pb + 8);
            acc[j] = __builtin_amdgcn_wmma_f32_16x16x32_bf16(
                false, a.v, false, bfr.v, (short)0, acc[j], false, false);
        }
    }
#pragma unroll
    for (int j = 0; j < 4; j++) {
        int d = j * 16 + l15;
#pragma unroll
        for (int r = 0; r < 8; r++) {
            int m = w * 16 + r + half * 8;
            float v = acc[j][r] / nrm[m];
            Out[((size_t)b * Nn + mt * 64 + m) * Dd + h * DHh + d] = f2bf(v);
        }
    }
}

// ---------------------------------------------------------------- stage 6: exp map (in place)
__global__ __launch_bounds__(256) void expmap_kernel(float* __restrict__ y,
                                                     const float* __restrict__ curv) {
    const int t = blockIdx.x;
    float* row = y + (size_t)t * Dd;
    __shared__ float red[256];
    float ss = 0.f;
    for (int i = threadIdx.x; i < Dd; i += 256) { float v = row[i]; ss += v * v; }
    red[threadIdx.x] = ss; __syncthreads();
    for (int s = 128; s > 0; s >>= 1) {
        if (threadIdx.x < s) red[threadIdx.x] += red[threadIdx.x + s];
        __syncthreads();
    }
    float c  = fmaxf(fabsf(curv[0]), 1e-6f);
    float sc = sqrtf(fmaxf(c, 1e-6f));
    float vn = fminf(fmaxf(sqrtf(red[0]), 1e-6f), 15.f);
    float targ = fminf(sc * vn, 15.f);
    float fac = tanhf(targ) / (sc * vn + 1e-8f);
    for (int i = threadIdx.x; i < Dd; i += 256) row[i] *= fac;
}

// ---------------------------------------------------------------- launch
extern "C" void kernel_launch(void* const* d_in, const int* in_sizes, int n_in,
                              void* d_out, int out_size, void* d_ws, size_t ws_size,
                              hipStream_t stream) {
    (void)in_sizes; (void)n_in; (void)out_size; (void)ws_size;
    const float* x    = (const float*)d_in[0];
    const float* curv = (const float*)d_in[1];
    const float* Wq = (const float*)d_in[2]; const float* bq = (const float*)d_in[3];
    const float* Wk = (const float*)d_in[4]; const float* bk = (const float*)d_in[5];
    const float* Wv = (const float*)d_in[6]; const float* bv = (const float*)d_in[7];
    const float* Wo = (const float*)d_in[8]; const float* bo = (const float*)d_in[9];

    const size_t DDn = (size_t)Dd * Dd;           // 1M elems
    const size_t MB  = 1024ull * 1024ull;
    char* ws = (char*)d_ws;
    unsigned short* Wq_bf = (unsigned short*)(ws + 0 * MB);
    unsigned short* Wk_bf = (unsigned short*)(ws + 2 * MB);
    unsigned short* Wv_bf = (unsigned short*)(ws + 4 * MB);
    unsigned short* Wo_bf = (unsigned short*)(ws + 6 * MB);
    unsigned short* xt    = (unsigned short*)(ws + 8 * MB);    // 32MB; reused as attn out
    unsigned short* qb    = (unsigned short*)(ws + 40 * MB);   // 32MB (phi_q)
    unsigned short* kb    = (unsigned short*)(ws + 72 * MB);   // 32MB (phi_k)
    unsigned short* vb    = (unsigned short*)(ws + 104 * MB);  // 32MB
    float*          kvp   = (float*)(ws + 136 * MB);           // 8MB  (kv partials, 8 chunks)
    float*          kspp  = (float*)(ws + 144 * MB);           // 128KB (ksum partials)
    unsigned short* kvb   = (unsigned short*)(ws + 145 * MB);  // 512KB (kv bf16)
    float*          ksumf = (float*)(ws + 146 * MB);           // 16KB

    const int Mtok = Bb * Nn;                      // 16384

    // 0) weights -> bf16
    int wgrid = (int)((DDn + 255) / 256);
    cvt_bf16_kernel<<<wgrid, 256, 0, stream>>>(Wq, Wq_bf, (int)DDn);
    cvt_bf16_kernel<<<wgrid, 256, 0, stream>>>(Wk, Wk_bf, (int)DDn);
    cvt_bf16_kernel<<<wgrid, 256, 0, stream>>>(Wv, Wv_bf, (int)DDn);
    cvt_bf16_kernel<<<wgrid, 256, 0, stream>>>(Wo, Wo_bf, (int)DDn);

    // 1) log map -> x_tan (bf16)
    logmap_kernel<<<Mtok, 256, 0, stream>>>(x, curv, xt);

    // 2) Q/K/V projections (Q,K get elu+1 fused)
    dim3 ggrid(Dd / 128, Mtok / 128);
    gemm_bf16_kernel<0><<<ggrid, 256, 0, stream>>>(xt, Wq_bf, bq, qb, nullptr, Mtok, Dd, Dd);
    gemm_bf16_kernel<0><<<ggrid, 256, 0, stream>>>(xt, Wk_bf, bk, kb, nullptr, Mtok, Dd, Dd);
    gemm_bf16_kernel<1><<<ggrid, 256, 0, stream>>>(xt, Wv_bf, bv, vb, nullptr, Mtok, Dd, Dd);

    // 3) split-K kv + k_sum partials (deterministic), then reduce
    ksum_split_kernel<<<dim3(Bb * Hh, NCHUNK), 64, 0, stream>>>(kb, kspp);
    kv_split_kernel<<<dim3(Bb * Hh, NCHUNK), 128, 0, stream>>>(kb, vb, kvp);
    {
        int ntot = Bb * Hh * DHh * DHh + Bb * Hh * DHh;
        kv_reduce_kernel<<<(ntot + 255) / 256, 256, 0, stream>>>(kvp, kspp, kvb, ksumf);
    }

    // 4) qkv + normalize -> attn (reuses xt buffer)
    qkv_kernel<<<dim3(Bb * Hh, Nn / 64), 128, 0, stream>>>(qb, kvb, ksumf, xt);

    // 5) output projection -> f32 into d_out
    gemm_bf16_kernel<2><<<ggrid, 256, 0, stream>>>(xt, Wo_bf, bo, nullptr, (float*)d_out,
                                                   Mtok, Dd, Dd);

    // 6) exp map in place
    expmap_kernel<<<Mtok, 256, 0, stream>>>((float*)d_out, curv);
}
